// AGAT_38637525795123
// MI455X (gfx1250) — matrix operand order repaired and checked by reading, not compiled, not run on hardware
//
#include <hip/hip_runtime.h>

#define Bq 8
#define Nq 2048
#define FINq 256
#define FOUTq 256
#define ALPHAq 0.2f
#define NEG_BIGq (-9.0e15f)
#define SEGq 8

typedef __attribute__((ext_vector_type(16))) __bf16 v16bf;
typedef __attribute__((ext_vector_type(8)))  __bf16 v8bf;
typedef __attribute__((ext_vector_type(4)))  __bf16 v4bf;
typedef __attribute__((ext_vector_type(8)))  float  v8f;

__device__ __forceinline__ v16bf cat8(v8bf lo, v8bf hi) {
  return __builtin_shufflevector(lo, hi, 0,1,2,3,4,5,6,7,8,9,10,11,12,13,14,15);
}

// Async copy 16B global -> LDS (per active lane), tracked by ASYNCcnt.
__device__ __forceinline__ void async_g2l_b128(unsigned int lds_addr, const void* gptr) {
  asm volatile("global_load_async_to_lds_b128 %0, %1, off"
               :: "v"(lds_addr), "v"((unsigned long long)(size_t)gptr)
               : "memory");
}
__device__ __forceinline__ void wait_asynccnt0() {
  asm volatile("s_wait_asynccnt 0x0" ::: "memory");
}

// ---------------- prep: WbfT[o][f] = bf16(W[f][o]) ----------------
__global__ __launch_bounds__(256) void k_prep_w(const float* __restrict__ W,
                                                __bf16* __restrict__ WbfT) {
  const int idx = blockIdx.x * 256 + threadIdx.x;   // over FOUT*FIN
  const int o = idx >> 8;
  const int f = idx & 255;
  WbfT[idx] = (__bf16)W[f * FOUTq + o];
}

// ---------------- prep: wa1[f]=sum_o W[f][o]a1[o], wa2 likewise ----------------
__global__ __launch_bounds__(256) void k_prep_wa(const float* __restrict__ W,
                                                 const float* __restrict__ a,
                                                 float* __restrict__ wa) {
  const int f = threadIdx.x;                        // one block of 256
  float s1 = 0.f, s2 = 0.f;
  const float* wr = W + f * FOUTq;
  for (int o = 0; o < FOUTq; ++o) {
    s1 += wr[o] * a[o];
    s2 += wr[o] * a[FOUTq + o];
  }
  wa[f] = s1;
  wa[FINq + f] = s2;
}

// ---------------- prep: xbf = bf16(x), vectorized ----------------
__global__ __launch_bounds__(256) void k_prep_x(const float* __restrict__ x,
                                                __bf16* __restrict__ xbf) {
  const int idx = (blockIdx.x * 256 + threadIdx.x) * 4;
  const float4 v = *(const float4*)(x + idx);
  v4bf o = { (__bf16)v.x, (__bf16)v.y, (__bf16)v.z, (__bf16)v.w };
  *(v4bf*)(xbf + idx) = o;
}

// ---------------- s1[b,i] = x_row . wa1 ; s2 likewise (fp32) ----------------
__global__ __launch_bounds__(256) void k_s12(const float* __restrict__ x,
                                             const float* __restrict__ wa,
                                             float* __restrict__ s1g,
                                             float* __restrict__ s2g) {
  __shared__ float sw1[FINq];
  __shared__ float sw2[FINq];
  const int tid = threadIdx.x;
  sw1[tid] = wa[tid];
  sw2[tid] = wa[FINq + tid];
  __syncthreads();
  const int g = blockIdx.x * 256 + tid;             // over B*N
  const float* xp = x + (size_t)g * FINq;
  float s1 = 0.f, s2 = 0.f;
  for (int f = 0; f < FINq; f += 4) {
    const float4 xv = *(const float4*)(xp + f);
    s1 += xv.x * sw1[f] + xv.y * sw1[f + 1] + xv.z * sw1[f + 2] + xv.w * sw1[f + 3];
    s2 += xv.x * sw2[f] + xv.y * sw2[f + 1] + xv.z * sw2[f + 2] + xv.w * sw2[f + 3];
  }
  s1g[g] = s1;
  s2g[g] = s2;
}

// ---------------- h = x@W -> hbfT[b][o][n], A tile via async-to-LDS ----------------
__global__ __launch_bounds__(256) void k_gemm_h(const __bf16* __restrict__ xbf,
                                                const __bf16* __restrict__ WbfT,
                                                __bf16* __restrict__ hbfT) {
  const int b  = blockIdx.x >> 7;
  const int i0 = (blockIdx.x & 127) << 4;
  __shared__ __align__(16) __bf16 sA[16 * 40];      // 16 rows, 80B stride
  const int tid  = threadIdx.x;
  const int wave = tid >> 5;
  const int lane = tid & 31;
  const int n16  = lane & 15;
  const bool lo16 = lane < 16;
  const int koffA = lo16 ? 0 : 16;                  // A: K 0..15 / 16..31
  const int koffB = lo16 ? 0 : 16;                  // B: K 0..15 / 16..31

  // async-copy lane mapping (threads 0..63): 16 rows x 4 x 16B
  const int arow = tid >> 2;
  const int aq   = tid & 3;
  const unsigned int lds_a =
      (unsigned int)(size_t)&sA[arow * 40] + (unsigned int)(aq * 16);
  const __bf16* gsrc = xbf + ((size_t)b * Nq + i0 + arow) * FINq + aq * 8;

  v8f acc[2] = {v8f{0.f,0.f,0.f,0.f,0.f,0.f,0.f,0.f},
                v8f{0.f,0.f,0.f,0.f,0.f,0.f,0.f,0.f}};

  for (int f0 = 0; f0 < FINq; f0 += 32) {
    if (tid < 64) {
      async_g2l_b128(lds_a, gsrc + f0);
      wait_asynccnt0();
    }
    __syncthreads();
    const __bf16* pa = &sA[n16 * 40 + koffA];
    v16bf afrag = cat8(*(const v8bf*)pa, *(const v8bf*)(pa + 8));
#pragma unroll
    for (int t = 0; t < 2; ++t) {
      const int col = ((wave << 1) + t) * 16 + n16;
      const __bf16* pb = WbfT + (size_t)col * FINq + f0 + koffB;
      v16bf bfrag = cat8(*(const v8bf*)pb, *(const v8bf*)(pb + 8));
      acc[t] = __builtin_amdgcn_wmma_f32_16x16x32_bf16(false, afrag, false, bfrag,
                                                       (short)0, acc[t], false, false);
    }
    __syncthreads();
  }
  const int rbase = lo16 ? 0 : 8;
#pragma unroll
  for (int t = 0; t < 2; ++t) {
    const int col = ((wave << 1) + t) * 16 + n16;
    v8bf o;
#pragma unroll
    for (int r = 0; r < 8; ++r) o[r] = (__bf16)acc[t][r];
    *(v8bf*)&hbfT[((size_t)b * FOUTq + col) * Nq + i0 + rbase] = o;
  }
}

// ---------------- per-segment online column max/sum over i ----------------
__global__ __launch_bounds__(256) void k_colpart(const int* __restrict__ adj,
                                                 const float* __restrict__ s1g,
                                                 const float* __restrict__ s2g,
                                                 float* __restrict__ pmax,
                                                 float* __restrict__ psum) {
  const int b    = blockIdx.x >> 6;
  const int seg  = (blockIdx.x >> 3) & 7;
  const int jblk = blockIdx.x & 7;
  const int j = (jblk << 8) + threadIdx.x;
  const float s2j = s2g[b * Nq + j];
  float m = -__builtin_inff();
  float ssum = 0.f;
  const int* ap = adj + (size_t)b * Nq * Nq + j;
  const int ibase = seg << 8;
  for (int ii = 0; ii < 256; ++ii) {
    const int i = ibase + ii;
    const int av = ap[(size_t)i * Nq];
    float e = s1g[b * Nq + i] + s2j;
    e = (e > 0.f) ? e : ALPHAq * e;
    const float v = (av > 0) ? e : NEG_BIGq;
    const float mn = fmaxf(m, v);
    ssum = ssum * __expf(m - mn) + __expf(v - mn);
    m = mn;
  }
  pmax[((size_t)seg * Bq + b) * Nq + j] = m;
  psum[((size_t)seg * Bq + b) * Nq + j] = ssum;
}

// ---------------- combine segments -> colmax, 1/colsum ----------------
__global__ __launch_bounds__(256) void k_colcomb(const float* __restrict__ pmax,
                                                 const float* __restrict__ psum,
                                                 float* __restrict__ cmax,
                                                 float* __restrict__ crcp) {
  const int g = blockIdx.x * 256 + threadIdx.x;
  const int b = g / Nq;
  const int j = g % Nq;
  float m = -__builtin_inff();
  float ssum = 0.f;
#pragma unroll
  for (int seg = 0; seg < SEGq; ++seg) {
    const float pm = pmax[((size_t)seg * Bq + b) * Nq + j];
    const float ps = psum[((size_t)seg * Bq + b) * Nq + j];
    const float mn = fmaxf(m, pm);
    ssum = ssum * __expf(m - mn) + ps * __expf(pm - mn);
    m = mn;
  }
  cmax[g] = m;
  crcp[g] = __frcp_rn(ssum);
}

// ---------------- h' = att @ h, fused softmax-apply + ELU ----------------
__global__ __launch_bounds__(256) void k_att(const int* __restrict__ adj,
                                             const __bf16* __restrict__ hbfT,
                                             const float* __restrict__ s1g,
                                             const float* __restrict__ s2g,
                                             const float* __restrict__ cmax,
                                             const float* __restrict__ crcp,
                                             float* __restrict__ out) {
  const int b  = blockIdx.x >> 7;
  const int i0 = (blockIdx.x & 127) << 4;
  __shared__ __align__(16) __bf16 sP[16 * 40];
  __shared__ float sS2[Nq];
  __shared__ float sCM[Nq];
  __shared__ float sCR[Nq];
  __shared__ float sS1[16];
  const int tid  = threadIdx.x;
  const int wave = tid >> 5;
  const int lane = tid & 31;
  const int n16  = lane & 15;
  const bool lo16 = lane < 16;
  const int koffA = lo16 ? 0 : 16;
  const int koffB = lo16 ? 0 : 16;

  for (int t = tid; t < Nq; t += 256) {
    sS2[t] = s2g[b * Nq + t];
    sCM[t] = cmax[b * Nq + t];
    sCR[t] = crcp[b * Nq + t];
  }
  if (tid < 16) sS1[tid] = s1g[b * Nq + i0 + tid];
  __syncthreads();

  v8f acc[2] = {v8f{0.f,0.f,0.f,0.f,0.f,0.f,0.f,0.f},
                v8f{0.f,0.f,0.f,0.f,0.f,0.f,0.f,0.f}};
  const size_t adjbase = (size_t)b * Nq * Nq;

  for (int j0 = 0; j0 < Nq; j0 += 32) {
    for (int e = tid; e < 512; e += 256) {
      const int r = e >> 5, k = e & 31;
      const int j = j0 + k;
      const int av = adj[adjbase + (size_t)(i0 + r) * Nq + j];
      if (j0 + 32 < Nq)
        __builtin_prefetch(&adj[adjbase + (size_t)(i0 + r) * Nq + j + 32], 0, 0);
      float ev = sS1[r] + sS2[j];
      ev = (ev > 0.f) ? ev : ALPHAq * ev;
      const float v = (av > 0) ? ev : NEG_BIGq;
      const float p = __expf(v - sCM[j]) * sCR[j];
      sP[r * 40 + k] = (__bf16)p;
    }
    __syncthreads();
    const __bf16* pa = &sP[n16 * 40 + koffA];
    v16bf afrag = cat8(*(const v8bf*)pa, *(const v8bf*)(pa + 8));
#pragma unroll
    for (int t = 0; t < 2; ++t) {
      const int col = ((wave << 1) + t) * 16 + n16;
      const __bf16* pb = hbfT + ((size_t)b * FOUTq + col) * Nq + j0 + koffB;
      v16bf bfrag = cat8(*(const v8bf*)pb, *(const v8bf*)(pb + 8));
      acc[t] = __builtin_amdgcn_wmma_f32_16x16x32_bf16(false, afrag, false, bfrag,
                                                       (short)0, acc[t], false, false);
    }
    __syncthreads();
  }

  const int rbase = lo16 ? 0 : 8;
#pragma unroll
  for (int t = 0; t < 2; ++t) {
    const int col = ((wave << 1) + t) * 16 + n16;
#pragma unroll
    for (int r = 0; r < 8; ++r) {
      const float v = acc[t][r];
      const float o = (v > 0.f) ? v : (__expf(v) - 1.0f);
      out[((size_t)b * Nq + i0 + rbase + r) * FOUTq + col] = o;
    }
  }
}

// ---------------- launcher ----------------
extern "C" void kernel_launch(void* const* d_in, const int* in_sizes, int n_in,
                              void* d_out, int out_size, void* d_ws, size_t ws_size,
                              hipStream_t stream) {
  const float* x   = (const float*)d_in[0];
  const int*   adj = (const int*)d_in[1];
  const float* W   = (const float*)d_in[2];
  const float* a   = (const float*)d_in[3];
  float* out = (float*)d_out;

  char* w = (char*)d_ws;
  __bf16* WbfT = (__bf16*)(w);                      // 131072
  __bf16* hbfT = (__bf16*)(w + 131072);             // 8388608
  __bf16* xbf  = (__bf16*)(w + 8519680);            // 8388608
  float*  s1g  = (float*)(w + 16908288);            // 65536
  float*  s2g  = (float*)(w + 16973824);
  float*  cmx  = (float*)(w + 17039360);
  float*  crp  = (float*)(w + 17104896);
  float*  pmax = (float*)(w + 17170432);            // 524288
  float*  psum = (float*)(w + 17694720);
  float*  wa   = (float*)(w + 18219008);            // 2048

  k_prep_w<<<(FOUTq * FINq) / 256, 256, 0, stream>>>(W, WbfT);
  k_prep_wa<<<1, 256, 0, stream>>>(W, a, wa);
  k_prep_x<<<(Bq * Nq * FINq) / 1024, 256, 0, stream>>>(x, xbf);
  k_s12<<<(Bq * Nq) / 256, 256, 0, stream>>>(x, wa, s1g, s2g);
  k_gemm_h<<<Bq * (Nq / 16), 256, 0, stream>>>(xbf, WbfT, hbfT);
  k_colpart<<<Bq * SEGq * (Nq / 256), 256, 0, stream>>>(adj, s1g, s2g, pmax, psum);
  k_colcomb<<<(Bq * Nq) / 256, 256, 0, stream>>>(pmax, psum, cmx, crp);
  k_att<<<Bq * (Nq / 16), 256, 0, stream>>>(adj, hbfT, s1g, s2g, cmx, crp, out);
}
